// SAFR_Block_68401649156831
// MI455X (gfx1250) — matrix operand order repaired and checked by reading, not compiled, not run on hardware
//
#include <hip/hip_runtime.h>
#include <stdint.h>

// ---------------- constants ----------------
constexpr int kDim   = 512;   // channels
constexpr int kHeads = 8;
constexpr int kHd    = 64;    // head dim
constexpr int kNB    = 196;   // spatial positions (14x14)
constexpr int kN     = 256;   // tokens (attention seq) / conv batch
constexpr int kRows  = kNB * kN;  // 50176 total rows

typedef __attribute__((ext_vector_type(16))) __bf16 v16bf;
typedef __attribute__((ext_vector_type(8)))  float  v8f;

union FragAB { uint32_t u[8]; v16bf v; };

// ---------------- scalar helpers ----------------
__device__ __forceinline__ uint16_t f2bf(float f) {
  uint32_t u = __float_as_uint(f);
  return (uint16_t)((u + 0x7FFFu + ((u >> 16) & 1u)) >> 16);  // RNE
}
__device__ __forceinline__ float bf2f(uint16_t h) {
  return __uint_as_float(((uint32_t)h) << 16);
}
__device__ __forceinline__ uint32_t pack2(float f0, float f1) {
  return (uint32_t)f2bf(f0) | ((uint32_t)f2bf(f1) << 16);
}
__device__ __forceinline__ float geluf(float x) {
  return 0.5f * x * (1.0f + erff(x * 0.70710678118654752f));
}
__device__ __forceinline__ float wave_sum(float v) {
#pragma unroll
  for (int o = 16; o > 0; o >>= 1) v += __shfl_xor(v, o, 32);
  return v;
}
__device__ __forceinline__ float wave_max(float v) {
#pragma unroll
  for (int o = 16; o > 0; o >>= 1) v = fmaxf(v, __shfl_xor(v, o, 32));
  return v;
}

// ---------------- WMMA fragment loads (wave32 bf16 16x16x32 layouts) ----------------
// A: 16x32 tile, row-major, stride lda. lanes 0-15 hold M=0..15; lanes>=16 add K+8.
// VGPR v: K = 2*(v&3) + 16*(v>>2); pairs contiguous -> two b128 loads per lane.
__device__ __forceinline__ void load_a(FragAB& f, const uint16_t* __restrict__ A,
                                       int lda, size_t m0, int kk, int lane) {
  const uint16_t* row = A + (m0 + (size_t)(lane & 15)) * (size_t)lda + kk + ((lane >> 4) << 3);
#pragma unroll
  for (int v = 0; v < 8; ++v)
    f.u[v] = *(const uint32_t*)(row + (((v & 3) << 1) + ((v >> 2) << 4)));
}
// B: 32x16 tile from K-major ("transposed") Bt[n][k], stride ldb.
// n = lane&15; lanes>=16 add K+8; VGPR v: K = 2*(v&3) + 16*(v>>2) (same as A).
__device__ __forceinline__ void load_b(FragAB& f, const uint16_t* __restrict__ Bt,
                                       int ldb, int n0, int kk, int lane) {
  const uint16_t* row = Bt + (size_t)(n0 + (lane & 15)) * (size_t)ldb + kk + ((lane >> 4) << 3);
#pragma unroll
  for (int v = 0; v < 8; ++v)
    f.u[v] = *(const uint32_t*)(row + (((v & 3) << 1) + ((v >> 2) << 4)));
}

#define BF16_WMMA(a, b, c) \
  __builtin_amdgcn_wmma_f32_16x16x32_bf16(false, (a).v, false, (b).v, (short)0, (c), false, false)

// ---------------- weight prep ----------------
__global__ void k_transpose_bf16(const float* __restrict__ w, uint16_t* __restrict__ wt,
                                 int K, int N) {
  size_t idx = (size_t)blockIdx.x * blockDim.x + threadIdx.x;
  if (idx >= (size_t)K * N) return;
  int n = (int)(idx % N);
  int k = (int)(idx / N);
  wt[(size_t)n * K + k] = f2bf(w[idx]);
}
// pconv_w (co,ci,3,3) -> Wt[co][(tap)*128 + ci]  (32-wide K step stays inside one tap)
__global__ void k_pack_pconv(const float* __restrict__ w, uint16_t* __restrict__ wt) {
  int idx = blockIdx.x * blockDim.x + threadIdx.x;
  if (idx >= 128 * 128 * 9) return;
  int co = idx / (128 * 9);
  int r  = idx % (128 * 9);
  int ci = r / 9;
  int d  = r % 9;
  wt[co * 1152 + d * 128 + ci] = f2bf(w[idx]);
}

// ---------------- layernorm: one wave per 512-wide row, bf16 out ----------------
__global__ void k_layernorm(const float* __restrict__ x, const float* __restrict__ gam,
                            const float* __restrict__ bet, uint16_t* __restrict__ out, int rows) {
  int wave = (int)(((size_t)blockIdx.x * blockDim.x + threadIdx.x) >> 5);
  int lane = threadIdx.x & 31;
  if (wave >= rows) return;
  const float* row = x + (size_t)wave * kDim;
  float vals[16];
  float s = 0.f;
#pragma unroll
  for (int j = 0; j < 16; ++j) { vals[j] = row[lane + 32 * j]; s += vals[j]; }
  float mu = wave_sum(s) * (1.f / 512.f);
  float var = 0.f;
#pragma unroll
  for (int j = 0; j < 16; ++j) { float d = vals[j] - mu; var += d * d; }
  var = wave_sum(var) * (1.f / 512.f);
  float inv = rsqrtf(var + 1e-5f);
  uint16_t* o = out + (size_t)wave * kDim;
#pragma unroll
  for (int j = 0; j < 16; ++j) {
    int c = lane + 32 * j;
    o[c] = f2bf((vals[j] - mu) * inv * gam[c] + bet[c]);
  }
}

// ---------------- QKV GEMM (50176x512 @ 512x1536), double-buffered, q/k/vT scatter ----------------
__global__ void k_gemm_qkv(const uint16_t* __restrict__ A, const uint16_t* __restrict__ Bt,
                           uint16_t* __restrict__ q, uint16_t* __restrict__ kmat,
                           uint16_t* __restrict__ vT) {
  const int M = kRows, N = 1536, K = 512;
  int lane = threadIdx.x & 31;
  int gw = (int)(((size_t)blockIdx.x * blockDim.x + threadIdx.x) >> 5);
  int mTiles = M >> 4, nQuads = N >> 6;
  if (gw >= mTiles * nQuads) return;
  int mt = gw % mTiles, nq = gw / mTiles;
  size_t m0 = (size_t)mt << 4;
  int n0 = nq << 6;
  v8f acc[4];
#pragma unroll
  for (int t = 0; t < 4; ++t)
#pragma unroll
    for (int e = 0; e < 8; ++e) acc[t][e] = 0.f;
  const int steps = K >> 5;  // 16 (even)
  FragAB aA, aB, b0[4], b1[4];
  load_a(aA, A, K, m0, 0, lane);
#pragma unroll
  for (int t = 0; t < 4; ++t) load_b(b0[t], Bt, K, n0 + t * 16, 0, lane);
  for (int s2 = 0; s2 < steps; s2 += 2) {
    int kOdd = (s2 + 1) << 5;
    load_a(aB, A, K, m0, kOdd, lane);
#pragma unroll
    for (int t = 0; t < 4; ++t) load_b(b1[t], Bt, K, n0 + t * 16, kOdd, lane);
#pragma unroll
    for (int t = 0; t < 4; ++t) acc[t] = BF16_WMMA(aA, b0[t], acc[t]);
    if (s2 + 2 < steps) {
      int kEven = (s2 + 2) << 5;
      load_a(aA, A, K, m0, kEven, lane);
#pragma unroll
      for (int t = 0; t < 4; ++t) load_b(b0[t], Bt, K, n0 + t * 16, kEven, lane);
    }
#pragma unroll
    for (int t = 0; t < 4; ++t) acc[t] = BF16_WMMA(aB, b1[t], acc[t]);
  }
#pragma unroll
  for (int t = 0; t < 4; ++t) {
    int c = n0 + t * 16 + (lane & 15);
#pragma unroll
    for (int v = 0; v < 8; ++v) {
      size_t r = m0 + v + ((lane >> 4) << 3);
      int bb = (int)(r >> 8), nn = (int)(r & 255);
      uint16_t val = f2bf(acc[t][v]);
      if (c < 512) {
        int h = c >> 6, d = c & 63;
        q[(((size_t)(bb * kHeads + h)) * kN + nn) * kHd + d] = val;
      } else if (c < 1024) {
        int c2 = c - 512, h = c2 >> 6, d = c2 & 63;
        kmat[(((size_t)(bb * kHeads + h)) * kN + nn) * kHd + d] = val;
      } else {
        int c2 = c - 1024, h = c2 >> 6, d = c2 & 63;
        vT[(((size_t)(bb * kHeads + h)) * kHd + d) * kN + nn] = val;  // pre-transposed V
      }
    }
  }
}

// ---------------- mask: sigmoid(mean_h(scores[b,h,last,m])) ----------------
__global__ void k_mask(const uint16_t* __restrict__ q, const uint16_t* __restrict__ k,
                       float* __restrict__ mask) {
  int b = blockIdx.x;   // 196
  int m = threadIdx.x;  // 256
  float acc = 0.f;
  for (int h = 0; h < kHeads; ++h) {
    const uint16_t* qr = q + (((size_t)(b * kHeads + h)) * kN + (kN - 1)) * kHd;
    const uint16_t* kr = k + (((size_t)(b * kHeads + h)) * kN + m) * kHd;
#pragma unroll 8
    for (int d = 0; d < kHd; ++d) acc += bf2f(qr[d]) * bf2f(kr[d]);
  }
  float v = acc * (0.125f * 0.125f);  // * scale(1/8) / heads(8)
  mask[b * kN + m] = 1.f / (1.f + expf(-v));
}

// ---------------- attention: full 256x256 score tile in 256KB LDS ----------------
// Phase 2 repacks softmax output in place as bf16 pairs (first half of each LDS row),
// so phase 3's A fragments are direct dword LDS reads (no f32->bf16 VALU).
__global__ void k_attn(const uint16_t* __restrict__ q, const uint16_t* __restrict__ kmat,
                       const uint16_t* __restrict__ vT, const float* __restrict__ mask,
                       uint16_t* __restrict__ obuf) {
  extern __shared__ float S[];  // 256*256 f32 = 256 KB (CDNA5 WGP has 320 KB LDS)
  int bh = blockIdx.x;
  int b = bh >> 3, h = bh & 7;
  int lane = threadIdx.x & 31;
  int wave = threadIdx.x >> 5;  // 8 waves
  const uint16_t* qb = q    + (size_t)bh * kN * kHd;
  const uint16_t* kb = kmat + (size_t)bh * kN * kHd;
  // phase 1: S = (q @ k^T) * scale  (K=64 -> both steps issued before the two WMMAs)
  for (int tt = 0; tt < 32; ++tt) {
    int t = wave * 32 + tt;
    int mt = t >> 4, nt = t & 15;
    v8f acc;
#pragma unroll
    for (int e = 0; e < 8; ++e) acc[e] = 0.f;
    FragAB a0, a1, bf0, bf1;
    load_a(a0, qb, kHd, (size_t)mt * 16, 0, lane);
    load_b(bf0, kb, kHd, nt * 16, 0, lane);
    load_a(a1, qb, kHd, (size_t)mt * 16, 32, lane);
    load_b(bf1, kb, kHd, nt * 16, 32, lane);
    acc = BF16_WMMA(a0, bf0, acc);
    acc = BF16_WMMA(a1, bf1, acc);
    int col = nt * 16 + (lane & 15);
#pragma unroll
    for (int v = 0; v < 8; ++v) {
      int m = mt * 16 + v + ((lane >> 4) << 3);
      S[m * kN + col] = acc[v] * 0.125f;
    }
  }
  __syncthreads();
  // phase 2: per-row softmax * mask; lane owns 8 contiguous cols; repack to bf16 in place
  float mk[8];
#pragma unroll
  for (int j = 0; j < 8; ++j) mk[j] = mask[b * kN + lane * 8 + j];
  for (int rr = 0; rr < 32; ++rr) {
    int r = wave * 32 + rr;
    float e[8], mx = -3.0e38f;
#pragma unroll
    for (int j = 0; j < 8; ++j) { e[j] = S[r * kN + lane * 8 + j]; mx = fmaxf(mx, e[j]); }
    mx = wave_max(mx);
    float sm = 0.f;
#pragma unroll
    for (int j = 0; j < 8; ++j) { e[j] = expf(e[j] - mx); sm += e[j]; }
    sm = wave_sum(sm);
    float inv = 1.f / sm;
    uint32_t* Pr = (uint32_t*)(S + r * kN);  // packed bf16 pairs, 128 dwords per row
#pragma unroll
    for (int t = 0; t < 4; ++t)
      Pr[lane * 4 + t] = pack2(e[2 * t] * inv * mk[2 * t], e[2 * t + 1] * inv * mk[2 * t + 1]);
  }
  __syncthreads();
  // phase 3: O = P @ V  (A = packed bf16 from LDS; vT is K-major Bt[d][m])
  const uint16_t* vb = vT + (size_t)bh * kHd * kN;
  for (int i = 0; i < 8; ++i) {
    int t = wave * 8 + i;
    int mt = t >> 2, nt = t & 3;
    v8f acc;
#pragma unroll
    for (int e2 = 0; e2 < 8; ++e2) acc[e2] = 0.f;
    int m = mt * 16 + (lane & 15);
    const uint32_t* Pr = (const uint32_t*)(S + m * kN);
    int khp = (lane >> 4) << 2;  // pair offset for upper half-wave (K+8 -> +4 pairs)
    for (int k0 = 0; k0 < kN; k0 += 64) {
      FragAB a0, a1, bf0, bf1;
#pragma unroll
      for (int v = 0; v < 8; ++v)
        a0.u[v] = Pr[(k0 >> 1) + (v & 3) + ((v >> 2) << 3) + khp];
#pragma unroll
      for (int v = 0; v < 8; ++v)
        a1.u[v] = Pr[((k0 + 32) >> 1) + (v & 3) + ((v >> 2) << 3) + khp];
      load_b(bf0, vb, kN, nt * 16, k0, lane);
      load_b(bf1, vb, kN, nt * 16, k0 + 32, lane);
      acc = BF16_WMMA(a0, bf0, acc);
      acc = BF16_WMMA(a1, bf1, acc);
    }
    int c = nt * 16 + (lane & 15);
#pragma unroll
    for (int v = 0; v < 8; ++v) {
      int mm = mt * 16 + v + ((lane >> 4) << 3);
      obuf[((size_t)b * kN + mm) * kDim + h * kHd + c] = f2bf(acc[v]);
    }
  }
}

// ---------------- generic GEMM + bias + f32 residual -> f32 out (double-buffered) ----------------
__global__ void k_gemm_bias_resid(const uint16_t* __restrict__ A, const uint16_t* __restrict__ Bt,
                                  const float* __restrict__ bias, const float* __restrict__ resid,
                                  float* __restrict__ out, int M, int N, int K) {
  int lane = threadIdx.x & 31;
  int gw = (int)(((size_t)blockIdx.x * blockDim.x + threadIdx.x) >> 5);
  int mTiles = M >> 4, nQuads = N >> 6;
  if (gw >= mTiles * nQuads) return;
  int mt = gw % mTiles, nq = gw / mTiles;
  size_t m0 = (size_t)mt << 4;
  int n0 = nq << 6;
  v8f acc[4];
#pragma unroll
  for (int t = 0; t < 4; ++t)
#pragma unroll
    for (int e = 0; e < 8; ++e) acc[t][e] = 0.f;
  const int steps = K >> 5;  // 16 or 4 (even)
  FragAB aA, aB, b0[4], b1[4];
  load_a(aA, A, K, m0, 0, lane);
#pragma unroll
  for (int t = 0; t < 4; ++t) load_b(b0[t], Bt, K, n0 + t * 16, 0, lane);
  for (int s2 = 0; s2 < steps; s2 += 2) {
    int kOdd = (s2 + 1) << 5;
    load_a(aB, A, K, m0, kOdd, lane);
#pragma unroll
    for (int t = 0; t < 4; ++t) load_b(b1[t], Bt, K, n0 + t * 16, kOdd, lane);
#pragma unroll
    for (int t = 0; t < 4; ++t) acc[t] = BF16_WMMA(aA, b0[t], acc[t]);
    if (s2 + 2 < steps) {
      int kEven = (s2 + 2) << 5;
      load_a(aA, A, K, m0, kEven, lane);
#pragma unroll
      for (int t = 0; t < 4; ++t) load_b(b0[t], Bt, K, n0 + t * 16, kEven, lane);
    }
#pragma unroll
    for (int t = 0; t < 4; ++t) acc[t] = BF16_WMMA(aB, b1[t], acc[t]);
  }
#pragma unroll
  for (int t = 0; t < 4; ++t) {
    int c = n0 + t * 16 + (lane & 15);
#pragma unroll
    for (int v = 0; v < 8; ++v) {
      size_t r = m0 + v + ((lane >> 4) << 3);
      out[r * N + c] = acc[t][v] + bias[c] + resid[r * N + c];
    }
  }
}

// ---------------- 3x3 conv on first 128 ch as implicit GEMM, K = 9*128, double-buffered --------
__global__ void k_conv_gemm(const uint16_t* __restrict__ y, const uint16_t* __restrict__ Wt,
                            uint16_t* __restrict__ y2) {
  const int M = kRows, N = 128;
  int lane = threadIdx.x & 31;
  int gw = (int)(((size_t)blockIdx.x * blockDim.x + threadIdx.x) >> 5);
  int mTiles = M >> 4, nQuads = N >> 6;  // 3136, 2
  if (gw >= mTiles * nQuads) return;
  int mt = gw % mTiles, nq = gw / mTiles;
  int n0 = nq << 6;
  // m ordered (b2, s): m = b2*196 + s, s = 14*i + j
  int m  = (mt << 4) + (lane & 15);
  int b2 = m / 196;
  int s  = m - b2 * 196;
  int si = s / 14, sj = s - si * 14;
  int kh = (lane >> 4) << 3;
  v8f acc[4];
#pragma unroll
  for (int t = 0; t < 4; ++t)
#pragma unroll
    for (int e = 0; e < 8; ++e) acc[t][e] = 0.f;
  // flattened K step ks in [0,36): tap d = ks>>2, ci offset = (ks&3)*32; weight k0 = 32*ks
  auto loadAstep = [&](FragAB& a, int ks) {
    int d  = ks >> 2;
    int kc = (ks & 3) << 5;
    int di = d / 3 - 1, dj = d - (d / 3) * 3 - 1;
    int ii = si + di, jj = sj + dj;
    bool ok = ((unsigned)ii < 14u) && ((unsigned)jj < 14u);
    if (ok) {
      const uint16_t* p = y + ((size_t)((ii * 14 + jj) * kN + b2)) * kDim + kc + kh;
#pragma unroll
      for (int v = 0; v < 8; ++v)
        a.u[v] = *(const uint32_t*)(p + (((v & 3) << 1) + ((v >> 2) << 4)));
    } else {
#pragma unroll
      for (int v = 0; v < 8; ++v) a.u[v] = 0u;
    }
  };
  FragAB aA, aB, b0[4], b1[4];
  loadAstep(aA, 0);
#pragma unroll
  for (int t = 0; t < 4; ++t) load_b(b0[t], Wt, 1152, n0 + t * 16, 0, lane);
  for (int s2 = 0; s2 < 36; s2 += 2) {
    int kOdd = (s2 + 1) << 5;
    loadAstep(aB, s2 + 1);
#pragma unroll
    for (int t = 0; t < 4; ++t) load_b(b1[t], Wt, 1152, n0 + t * 16, kOdd, lane);
#pragma unroll
    for (int t = 0; t < 4; ++t) acc[t] = BF16_WMMA(aA, b0[t], acc[t]);
    if (s2 + 2 < 36) {
      int kEven = (s2 + 2) << 5;
      loadAstep(aA, s2 + 2);
#pragma unroll
      for (int t = 0; t < 4; ++t) load_b(b0[t], Wt, 1152, n0 + t * 16, kEven, lane);
    }
#pragma unroll
    for (int t = 0; t < 4; ++t) acc[t] = BF16_WMMA(aB, b1[t], acc[t]);
  }
#pragma unroll
  for (int t = 0; t < 4; ++t) {
    int c = n0 + t * 16 + (lane & 15);
#pragma unroll
    for (int v = 0; v < 8; ++v) {
      int mm = (mt << 4) + v + ((lane >> 4) << 3);
      int b2p = mm / 196;
      int sp  = mm - b2p * 196;
      y2[((size_t)sp * kN + b2p) * 128 + c] = f2bf(acc[t][v]);
    }
  }
}

// ---------------- lin1: split A (cols<128 from conv out, else LN2 out) + GELU -----------------
__global__ void k_gemm_lin1(const uint16_t* __restrict__ y2, const uint16_t* __restrict__ y,
                            const uint16_t* __restrict__ Bt, const float* __restrict__ bias,
                            uint16_t* __restrict__ g) {
  const int M = kRows, N = 256, K = 512;
  int lane = threadIdx.x & 31;
  int gw = (int)(((size_t)blockIdx.x * blockDim.x + threadIdx.x) >> 5);
  int mTiles = M >> 4, nQuads = N >> 6;
  if (gw >= mTiles * nQuads) return;
  int mt = gw % mTiles, nq = gw / mTiles;
  size_t m0 = (size_t)mt << 4;
  int n0 = nq << 6;
  v8f acc[4];
#pragma unroll
  for (int t = 0; t < 4; ++t)
#pragma unroll
    for (int e = 0; e < 8; ++e) acc[t][e] = 0.f;
  auto loadAsplit = [&](FragAB& a, int k0) {
    if (k0 < 128) load_a(a, y2, 128, m0, k0, lane);
    else          load_a(a, y,  512, m0, k0, lane);
  };
  const int steps = K >> 5;  // 16
  FragAB aA, aB, b0[4], b1[4];
  loadAsplit(aA, 0);
#pragma unroll
  for (int t = 0; t < 4; ++t) load_b(b0[t], Bt, 512, n0 + t * 16, 0, lane);
  for (int s2 = 0; s2 < steps; s2 += 2) {
    int kOdd = (s2 + 1) << 5;
    loadAsplit(aB, kOdd);
#pragma unroll
    for (int t = 0; t < 4; ++t) load_b(b1[t], Bt, 512, n0 + t * 16, kOdd, lane);
#pragma unroll
    for (int t = 0; t < 4; ++t) acc[t] = BF16_WMMA(aA, b0[t], acc[t]);
    if (s2 + 2 < steps) {
      int kEven = (s2 + 2) << 5;
      loadAsplit(aA, kEven);
#pragma unroll
      for (int t = 0; t < 4; ++t) load_b(b0[t], Bt, 512, n0 + t * 16, kEven, lane);
    }
#pragma unroll
    for (int t = 0; t < 4; ++t) acc[t] = BF16_WMMA(aB, b1[t], acc[t]);
  }
#pragma unroll
  for (int t = 0; t < 4; ++t) {
    int c = n0 + t * 16 + (lane & 15);
#pragma unroll
    for (int v = 0; v < 8; ++v) {
      size_t r = m0 + v + ((lane >> 4) << 3);
      g[r * 256 + c] = f2bf(geluf(acc[t][v] + bias[c]));
    }
  }
}

// ---------------- depthwise 3x3 + GELU + gate ----------------
__global__ void k_dwconv_gate(const uint16_t* __restrict__ g, const float* __restrict__ w,
                              const float* __restrict__ bias, uint16_t* __restrict__ gated) {
  size_t idx = (size_t)blockIdx.x * blockDim.x + threadIdx.x;
  if (idx >= (size_t)kRows * 128) return;
  int c = (int)(idx & 127);
  int r = (int)(idx >> 7);
  int s = r >> 8, b2 = r & 255;
  int si = s / 14, sj = s - si * 14;
  float acc = bias[c];
#pragma unroll
  for (int d = 0; d < 9; ++d) {
    int di = d / 3 - 1, dj = d % 3 - 1;
    int ii = si + di, jj = sj + dj;
    if ((unsigned)ii < 14u && (unsigned)jj < 14u) {
      int rp = (ii * 14 + jj) * kN + b2;
      acc += bf2f(g[(size_t)rp * 256 + c]) * w[c * 9 + d];
    }
  }
  float z  = geluf(acc);
  float g2 = bf2f(g[(size_t)r * 256 + 128 + c]);
  gated[(size_t)r * 128 + c] = f2bf(z * g2);
}

// ---------------- host ----------------
extern "C" void kernel_launch(void* const* d_in, const int* in_sizes, int n_in,
                              void* d_out, int out_size, void* d_ws, size_t ws_size,
                              hipStream_t stream) {
  (void)in_sizes; (void)n_in; (void)out_size; (void)ws_size;
  const float* x      = (const float*)d_in[0];
  const float* ln1_g  = (const float*)d_in[1];
  const float* ln1_b  = (const float*)d_in[2];
  const float* qkv_w  = (const float*)d_in[3];
  const float* proj_w = (const float*)d_in[4];
  const float* proj_b = (const float*)d_in[5];
  const float* ln2_g  = (const float*)d_in[6];
  const float* ln2_b  = (const float*)d_in[7];
  const float* pconv_w= (const float*)d_in[8];
  const float* lin1_w = (const float*)d_in[9];
  const float* lin1_b = (const float*)d_in[10];
  const float* dw_w   = (const float*)d_in[11];
  const float* dw_b   = (const float*)d_in[12];
  const float* lin2_w = (const float*)d_in[13];
  const float* lin2_b = (const float*)d_in[14];
  float* out = (float*)d_out;

  char* ws = (char*)d_ws;
  size_t off = 0;
  auto take = [&](size_t bytes) -> void* {
    void* p = ws + off;
    off += (bytes + 255) & ~(size_t)255;
    return p;
  };
  uint16_t* hB    = (uint16_t*)take((size_t)kRows * 512 * 2);  // LN1 out; reused as attn O
  uint16_t* qbuf  = (uint16_t*)take((size_t)kRows * 512 * 2);
  uint16_t* kbuf  = (uint16_t*)take((size_t)kRows * 512 * 2);
  uint16_t* vtbuf = (uint16_t*)take((size_t)kRows * 512 * 2);
  float*    maskp = (float*)   take((size_t)kNB * kN * 4);
  float*    x1    = (float*)   take((size_t)kRows * 512 * 4);
  uint16_t* ybuf  = (uint16_t*)take((size_t)kRows * 512 * 2);
  uint16_t* y2buf = (uint16_t*)take((size_t)kRows * 128 * 2);
  uint16_t* gbuf  = (uint16_t*)take((size_t)kRows * 256 * 2);
  uint16_t* gated = (uint16_t*)take((size_t)kRows * 128 * 2);
  uint16_t* qkvT  = (uint16_t*)take((size_t)1536 * 512 * 2);
  uint16_t* projT = (uint16_t*)take((size_t)512 * 512 * 2);
  uint16_t* lin1T = (uint16_t*)take((size_t)256 * 512 * 2);
  uint16_t* lin2T = (uint16_t*)take((size_t)512 * 128 * 2);
  uint16_t* pconvT= (uint16_t*)take((size_t)128 * 1152 * 2);
  uint16_t* obuf  = hB;  // alias: hB dead after QKV GEMM

  dim3 blk(256);
  // weight prep
  k_transpose_bf16<<<dim3((512 * 1536 + 255) / 256), blk, 0, stream>>>(qkv_w,  qkvT, 512, 1536);
  k_transpose_bf16<<<dim3((512 * 512  + 255) / 256), blk, 0, stream>>>(proj_w, projT, 512, 512);
  k_transpose_bf16<<<dim3((512 * 256  + 255) / 256), blk, 0, stream>>>(lin1_w, lin1T, 512, 256);
  k_transpose_bf16<<<dim3((128 * 512  + 255) / 256), blk, 0, stream>>>(lin2_w, lin2T, 128, 512);
  k_pack_pconv   <<<dim3((128 * 128 * 9 + 255) / 256), blk, 0, stream>>>(pconv_w, pconvT);

  // 1) LN1 -> bf16
  k_layernorm<<<dim3(kRows / 8), blk, 0, stream>>>(x, ln1_g, ln1_b, hB, kRows);
  // 2) QKV GEMM with scatter (q, k row-major per head; v pre-transposed)
  k_gemm_qkv<<<dim3((kRows / 16) * (1536 / 64) / 8), blk, 0, stream>>>(hB, qkvT, qbuf, kbuf, vtbuf);
  // 3) key-position mask from last query row
  k_mask<<<dim3(kNB), blk, 0, stream>>>(qbuf, kbuf, maskp);
  // 4) attention with 256 KB LDS score tile per (b,h)
  k_attn<<<dim3(kNB * kHeads), blk, 256 * 256 * sizeof(float), stream>>>(qbuf, kbuf, vtbuf, maskp, obuf);
  // 5) proj + bias + residual(x) -> x1 (f32)
  k_gemm_bias_resid<<<dim3((kRows / 16) * (512 / 64) / 8), blk, 0, stream>>>(
      obuf, projT, proj_b, x, x1, kRows, 512, 512);
  // 6) LN2 -> bf16
  k_layernorm<<<dim3(kRows / 8), blk, 0, stream>>>(x1, ln2_g, ln2_b, ybuf, kRows);
  // 7) 3x3 conv on first 128 channels (implicit GEMM)
  k_conv_gemm<<<dim3((kRows / 16) * (128 / 64) / 8), blk, 0, stream>>>(ybuf, pconvT, y2buf);
  // 8) lin1 + GELU (A cols<128 from conv output, rest from LN2 output)
  k_gemm_lin1<<<dim3((kRows / 16) * (256 / 64) / 8), blk, 0, stream>>>(y2buf, ybuf, lin1T, lin1_b, gbuf);
  // 9) depthwise 3x3 + GELU + gate
  k_dwconv_gate<<<dim3((kRows * 128) / 256), blk, 0, stream>>>(gbuf, dw_w, dw_b, gated);
  // 10) lin2 + bias + residual(x1) -> out (f32)
  k_gemm_bias_resid<<<dim3((kRows / 16) * (512 / 64) / 8), blk, 0, stream>>>(
      gated, lin2T, lin2_b, x1, out, kRows, 512, 128);
}
